// BayesGRU_10007273800415
// MI455X (gfx1250) — matrix operand order repaired and checked by reading, not compiled, not run on hardware
//
#include <hip/hip_runtime.h>

// ---------------------------------------------------------------------------
// BayesGRU (eval mode) for MI455X / gfx1250, wave32 + WMMA bf16.
//   T=512, B=64, D_IN=1024, D_H=1024
// Phase 0: pack all 6 weight matrices fp32 -> bf16 in WMMA B-fragment layout,
//          convert h0 to bf16, reset grid barrier.
// Phase 1: big parallel GEMM  Gx[t,gate,b,j] = emb[t] @ w_ih_gate^T  (fp32 out)
// Phase 2: persistent recurrent kernel, 512 steps, 2 grid barriers per step,
//          w_hh fragments resident in LDS (192 KB of the WGP's 320 KB).
// ---------------------------------------------------------------------------

#define T_STEPS  512
#define BATCH    64
#define DH       1024
#define BH       ((size_t)BATCH * DH)   // 65536
#define NBLK_REC 32                     // persistent blocks (<= WGP count)
#define THREADS  256                    // 8 wave32 per block

typedef __attribute__((ext_vector_type(16))) __bf16 v16bf;
typedef __attribute__((ext_vector_type(8)))  float  v8f;

union BFrag32 { uint4 q[2]; v16bf v; };

__device__ __forceinline__ v8f wmma_bf16(v16bf a, v16bf b, v8f c) {
  // emits v_wmma_f32_16x16x32_bf16
  return __builtin_amdgcn_wmma_f32_16x16x32_bf16(false, a, false, b,
                                                 (short)0, c, false, false);
}

// Packed B layout (built by pack_kernel):
//   elem((mat, nt, kc, lane, e)) = W[mat][n, k],
//   n = nt*16 + (lane&15),  k = kc*32 + (e&7) + ((e&8) ? 16 : 0) + ((lane&16) ? 8 : 0)
// matching the ISA 16-bit B (32x16) layout: lanes 0-15 hold K={0..7,16..23},
// lanes 16-31 hold K={8..15,24..31}, N = lane%16.
__device__ __forceinline__ v16bf load_packB(const __bf16* __restrict__ packB,
                                            unsigned mat, unsigned nt,
                                            unsigned kc, unsigned lane) {
  const uint4* p = (const uint4*)(packB +
      ((((size_t)mat * 64u + nt) * 32u + kc) * 512u) + (size_t)lane * 16u);
  BFrag32 f; f.q[0] = p[0]; f.q[1] = p[1];
  return f.v;
}

// Same fragment but from the LDS-resident copy: layout
//   lds[((matl*2 + ntl)*32 + kc)*512 + lane*16 + e]
__device__ __forceinline__ v16bf load_B_lds(const __bf16* smem,
                                            unsigned matl, unsigned ntl,
                                            unsigned kc, unsigned lane) {
  const uint4* p = (const uint4*)(smem +
      (((matl * 2u + ntl) * 32u + kc) * 512u) + (size_t)lane * 16u);
  BFrag32 f; f.q[0] = p[0]; f.q[1] = p[1];
  return f.v;
}

// ISA 16-bit A (16x32) layout: lane<16: K = {0..7, 16..23}; lane>=16: +8.
// Source is row-major [m][k] bf16 -> two contiguous 16B runs per lane.
__device__ __forceinline__ v16bf load_A_bf16(const __bf16* __restrict__ src,
                                             unsigned mbase, unsigned k0,
                                             unsigned lane) {
  unsigned m  = mbase + (lane & 15u);
  unsigned hb = (lane & 16u) ? 8u : 0u;
  const uint4* p = (const uint4*)(src + (size_t)m * DH + k0 + hb);
  BFrag32 f; f.q[0] = p[0]; f.q[1] = p[2];   // k0+hb..+7 and k0+hb+16..+23
  return f.v;
}

// Same A layout but source rows are fp32 (emb); convert to bf16 on the fly.
__device__ __forceinline__ v16bf load_A_f32(const float* __restrict__ src,
                                            unsigned mbase, unsigned k0,
                                            unsigned lane) {
  unsigned m  = mbase + (lane & 15u);
  unsigned hb = (lane & 16u) ? 8u : 0u;
  const float* p = src + (size_t)m * DH + k0 + hb;
  v16bf v;
#pragma unroll
  for (int i = 0; i < 8; ++i) v[i]     = (__bf16)p[i];
#pragma unroll
  for (int i = 0; i < 8; ++i) v[8 + i] = (__bf16)p[16 + i];
  return v;
}

// ---------------------------------------------------------------------------
// Phase 0: weight pack (6 x 1024 x 1024), h0 -> bf16, barrier reset.
// ---------------------------------------------------------------------------
__global__ void gru_pack_kernel(const float* __restrict__ wihu,
                                const float* __restrict__ wihr,
                                const float* __restrict__ wihc,
                                const float* __restrict__ whhu,
                                const float* __restrict__ whhr,
                                const float* __restrict__ whhc,
                                const float* __restrict__ hx,
                                __bf16* __restrict__ packB,
                                __bf16* __restrict__ hbuf,
                                unsigned* __restrict__ bar) {
  size_t idx = (size_t)blockIdx.x * 1024u + threadIdx.x;  // < 6*1024*1024
  unsigned e    = (unsigned)idx & 15u;
  unsigned lane = ((unsigned)idx >> 4) & 31u;
  unsigned kc   = ((unsigned)idx >> 9) & 31u;
  unsigned nt   = ((unsigned)idx >> 14) & 63u;
  unsigned mat  = (unsigned)(idx >> 20);
  unsigned n = nt * 16u + (lane & 15u);
  unsigned k = kc * 32u + (e & 7u) + ((e & 8u) ? 16u : 0u) + ((lane & 16u) ? 8u : 0u);
  const float* mats[6] = { wihu, wihr, wihc, whhu, whhr, whhc };
  packB[idx] = (__bf16)mats[mat][(size_t)n * DH + k];
  if (idx < BH) hbuf[idx] = (__bf16)hx[idx];
  if (idx == 0) *bar = 0u;
}

// ---------------------------------------------------------------------------
// Phase 1: Gx[t][gate][b][j] = emb[t] @ w_ih_gate^T   (no bias; fp32 out)
// One wave per (t, m_tile, n_tile); all 3 gate tiles share A-fragments.
// ---------------------------------------------------------------------------
__global__ void gru_input_gemm(const float* __restrict__ emb,
                               const __bf16* __restrict__ packB,
                               float* __restrict__ gx) {
  unsigned lane = threadIdx.x & 31u;
  unsigned wid  = blockIdx.x * 8u + (threadIdx.x >> 5);   // 0 .. 131071
  unsigned t    = wid >> 8;
  unsigned rem  = wid & 255u;
  unsigned mt   = rem >> 6;
  unsigned nt   = rem & 63u;

  const float* xrows = emb + (size_t)t * BH;
  v8f au = {}, ar = {}, ac = {};
#pragma unroll 2
  for (unsigned kc = 0; kc < 32; ++kc) {
    v16bf a  = load_A_f32(xrows, mt * 16u, kc * 32u, lane);
    v16bf b0 = load_packB(packB, 0, nt, kc, lane);
    v16bf b1 = load_packB(packB, 1, nt, kc, lane);
    v16bf b2 = load_packB(packB, 2, nt, kc, lane);
    au = wmma_bf16(a, b0, au);
    ar = wmma_bf16(a, b1, ar);
    ac = wmma_bf16(a, b2, ac);
  }
  // C/D layout: lane L, row r -> (m = 16*mt + r + (L>=16?8:0), n = 16*nt + L%16)
  unsigned n    = nt * 16u + (lane & 15u);
  unsigned mofs = mt * 16u + ((lane & 16u) ? 8u : 0u);
  size_t tb = (size_t)t * 3u * BH;
#pragma unroll
  for (int r = 0; r < 8; ++r) {
    size_t o = (size_t)(mofs + r) * DH + n;
    gx[tb + 0u * BH + o] = au[r];
    gx[tb + 1u * BH + o] = ar[r];
    gx[tb + 2u * BH + o] = ac[r];
  }
}

// ---------------------------------------------------------------------------
// Grid-wide barrier: monotonic counter, gen*NBLK target.
// ---------------------------------------------------------------------------
__device__ __forceinline__ void grid_barrier(unsigned* __restrict__ bar,
                                             unsigned& gen) {
  __threadfence();
  __syncthreads();
  if (threadIdx.x == 0) {
    unsigned target = (gen + 1u) * NBLK_REC;
    __hip_atomic_fetch_add(bar, 1u, __ATOMIC_ACQ_REL, __HIP_MEMORY_SCOPE_AGENT);
    while (__hip_atomic_load(bar, __ATOMIC_ACQUIRE, __HIP_MEMORY_SCOPE_AGENT) < target)
      __builtin_amdgcn_s_sleep(2);
  }
  __syncthreads();
  __threadfence();
  ++gen;
}

// ---------------------------------------------------------------------------
// Phase 2: persistent recurrent kernel. 32 blocks x 8 waves.
// Wave w of block blk owns tile (mt = w&3, nt = blk*2 + (w>>2)), so a block
// needs only 2 n-tile columns of the 3 w_hh matrices: 3*2*32KB = 192KB LDS,
// staged ONCE and reused for all 512 steps (WGP has 320KB LDS).
// ---------------------------------------------------------------------------
__global__ void __launch_bounds__(THREADS, 1)
gru_recurrent(const float* __restrict__ bu, const float* __restrict__ br,
              const float* __restrict__ bc, const float* __restrict__ gx,
              const __bf16* __restrict__ packB,
              __bf16* __restrict__ hbuf, __bf16* __restrict__ rhbuf,
              unsigned* __restrict__ bar, float* __restrict__ out) {
  __shared__ uint4 smem4[12288];                 // 192 KB (320 KB WGP LDS)
  __bf16* smem = (__bf16*)smem4;

  unsigned lane = threadIdx.x & 31u;
  unsigned w    = threadIdx.x >> 5;              // 0..7
  unsigned mt   = w & 3u;                        // 0..3
  unsigned ntl  = w >> 2;                        // 0..1
  unsigned nt   = blockIdx.x * 2u + ntl;         // 0..63
  unsigned n    = nt * 16u + (lane & 15u);
  unsigned mofs = mt * 16u + ((lane & 16u) ? 8u : 0u);

  // ---- stage w_hh_{u,r,c} fragments for nt0/nt1 into LDS (once) ----
  {
    uint4* dst = smem4;
    for (unsigned c = 0; c < 6; ++c) {           // c = matl*2 + ntl
      unsigned matl = c >> 1, nl = c & 1u;
      const uint4* src = (const uint4*)(packB +
          (((size_t)(3u + matl) * 64u + (blockIdx.x * 2u + nl)) * 32u) * 512u);
      for (unsigned i = threadIdx.x; i < 2048u; i += THREADS)
        dst[(size_t)c * 2048u + i] = src[i];
    }
    __syncthreads();
  }

  float vbu = bu[n], vbr = br[n], vbc = bc[n];   // biases hoisted
  unsigned gen = 0;
  float upd[8];

  for (unsigned t = 0; t < T_STEPS; ++t) {
    const float* gxt = gx + (size_t)t * 3u * BH;

    // Hoist step-invariant loads: all 3 gates' Gx and h_prev, so their
    // latency hides under the 32-iteration WMMA loop below.
    float gxu[8], gxr[8], gxc[8], hprev[8];
#pragma unroll
    for (int r = 0; r < 8; ++r) {
      size_t o = (size_t)(mofs + r) * DH + n;
      gxu[r]   = gxt[o];
      gxr[r]   = gxt[BH + o];
      gxc[r]   = gxt[2u * BH + o];
      hprev[r] = (float)hbuf[o];
    }

    // ---- phase A: u and r gates over h_{t-1} ----
    v8f accu = {}, accr = {};
#pragma unroll 2
    for (unsigned kc = 0; kc < 32; ++kc) {
      v16bf a   = load_A_bf16(hbuf, mt * 16u, kc * 32u, lane);
      v16bf bu_ = load_B_lds(smem, 0, ntl, kc, lane);
      v16bf br_ = load_B_lds(smem, 1, ntl, kc, lane);
      accu = wmma_bf16(a, bu_, accu);
      accr = wmma_bf16(a, br_, accr);
    }
#pragma unroll
    for (int r = 0; r < 8; ++r) {
      size_t o = (size_t)(mofs + r) * DH + n;
      float pu = accu[r] + gxu[r] + vbu;
      float pr = accr[r] + gxr[r] + vbr;
      float u  = 1.0f / (1.0f + __expf(-pu));
      float rg = 1.0f / (1.0f + __expf(-pr));
      upd[r]   = u;
      rhbuf[o] = (__bf16)(rg * hprev[r]);
    }
    grid_barrier(bar, gen);

    // ---- phase B: c gate over rh, then state update ----
    v8f accc = {};
#pragma unroll 2
    for (unsigned kc = 0; kc < 32; ++kc) {
      v16bf a   = load_A_bf16(rhbuf, mt * 16u, kc * 32u, lane);
      v16bf bc_ = load_B_lds(smem, 2, ntl, kc, lane);
      accc = wmma_bf16(a, bc_, accc);
    }
#pragma unroll
    for (int r = 0; r < 8; ++r) {
      size_t o  = (size_t)(mofs + r) * DH + n;
      float pc  = accc[r] + gxc[r] + vbc;
      float cel = tanhf(pc);
      float hn  = hprev[r] + upd[r] * (cel - hprev[r]);  // (1-u)h + u*c
      out[(size_t)t * BH + o] = hn;
      hbuf[o] = (__bf16)hn;
      if (t == T_STEPS - 1)
        out[(size_t)T_STEPS * BH + o] = hn;              // hx_final tail
    }
    grid_barrier(bar, gen);
  }
}

// ---------------------------------------------------------------------------
extern "C" void kernel_launch(void* const* d_in, const int* in_sizes, int n_in,
                              void* d_out, int out_size, void* d_ws, size_t ws_size,
                              hipStream_t stream) {
  (void)in_sizes; (void)n_in; (void)out_size; (void)ws_size;
  const float* emb  = (const float*)d_in[0];
  const float* hx   = (const float*)d_in[1];
  const float* wihu = (const float*)d_in[2];
  const float* wihr = (const float*)d_in[3];
  const float* wihc = (const float*)d_in[4];
  const float* whhu = (const float*)d_in[5];
  const float* whhr = (const float*)d_in[6];
  const float* whhc = (const float*)d_in[7];
  const float* bu   = (const float*)d_in[8];
  const float* br   = (const float*)d_in[9];
  const float* bc   = (const float*)d_in[10];

  char* ws = (char*)d_ws;
  unsigned* bar   = (unsigned*)(ws);                       // 256 B reserved
  __bf16*   hbuf  = (__bf16*)(ws + 256);                   // 128 KB
  __bf16*   rhbuf = (__bf16*)(ws + 256 + BH * 2);          // 128 KB
  __bf16*   packB = (__bf16*)(ws + 256 + 2 * BH * 2);      // 12 MB
  float*    gxbuf = (float*) (ws + 256 + 2 * BH * 2
                                  + (size_t)6 * DH * DH * 2); // 384 MB
  float* out = (float*)d_out;

  gru_pack_kernel<<<6144, 1024, 0, stream>>>(wihu, wihr, wihc,
                                             whhu, whhr, whhc,
                                             hx, packB, hbuf, bar);
  gru_input_gemm<<<16384, THREADS, 0, stream>>>(emb, packB, gxbuf);
  gru_recurrent<<<NBLK_REC, THREADS, 0, stream>>>(bu, br, bc, gxbuf, packB,
                                                  hbuf, rhbuf, bar, out);
}
